// myxLSTM_48438641164614
// MI455X (gfx1250) — matrix operand order repaired and checked by reading, not compiled, not run on hardware
//
#include <hip/hip_runtime.h>
#include <hip/hip_bf16.h>

// ---------------- model constants ----------------
#define B_    64
#define S_TOT 500
#define EMB_  256
#define NH_   4
#define DH_   64
#define UPD   384
#define NTOK  (B_*S_TOT)      // 32000
#define FEAT  6656            // 128*52 conv features per patch
#define LDA_T 6664            // padded LDS row stride (keeps 16B alignment, breaks bank conflicts)

typedef __bf16 bf16;
typedef bf16 v16bf __attribute__((ext_vector_type(16)));
typedef bf16 v8bf  __attribute__((ext_vector_type(8)));
typedef float v8f  __attribute__((ext_vector_type(8)));
typedef unsigned int u32x4 __attribute__((ext_vector_type(4)));
typedef unsigned int u32x8 __attribute__((ext_vector_type(8)));

// ---------------- WMMA fragment helpers (CDNA5 ISA §7.12.2) ----------------
// A 16x32 bf16: lane L holds row (L&15); lanes<16 K {kb+0..15}, lanes>=16 K {kb+16..31}.
// Two 16-byte loads per lane.
__device__ inline v16bf load_a_frag(const bf16* A, int lda, int kb, int lane){
  int half = lane >> 4;
  const bf16* p = A + (size_t)(lane & 15) * lda + kb + half*16;
  v8bf lo = *(const v8bf*)p;
  v8bf hi = *(const v8bf*)(p + 8);
  v16bf r;
  #pragma unroll
  for (int i = 0; i < 8; ++i){ r[i] = lo[i]; r[i+8] = hi[i]; }
  return r;
}
// B 32x16 bf16 from row-major (N,K) weights: lane L holds column (L&15);
// lanes<16 K {kb+0..15}, lanes>=16 K {kb+16..31}; two 16-byte loads per lane.
__device__ inline v16bf load_b_frag(const bf16* Bw, int ldb, int kb, int lane){
  int half = lane >> 4;
  const bf16* p = Bw + (size_t)(lane & 15) * ldb + kb + half*16;
  v8bf lo = *(const v8bf*)p;
  v8bf hi = *(const v8bf*)(p + 8);
  v16bf r;
  #pragma unroll
  for (int i = 0; i < 8; ++i){ r[i] = lo[i]; r[i+8] = hi[i]; }
  return r;
}
// LDS fragment load for the TDM-staged tile (ldb fixed at 32 elements). Plain loads:
// bsm gets a visible cooperative zero-init store, and the asm-volatile TDM issue
// ("memory" clobber) + barriers prevent any store-to-load forwarding across iterations,
// so these lower to wave-scope ds_load_b128 pairs.
__device__ inline v16bf load_b_frag_lds(const bf16* Bl, int lane){
  int half = lane >> 4;
  const v8bf* p = (const v8bf*)(Bl + (size_t)(lane & 15) * 32 + half*16);
  v8bf lo = p[0];
  v8bf hi = p[1];
  v16bf r;
  #pragma unroll
  for (int i = 0; i < 8; ++i){ r[i] = lo[i]; r[i+8] = hi[i]; }
  return r;
}
__device__ inline v8f wmma_bf16(v16bf a, v16bf b, v8f c){
  return __builtin_amdgcn_wmma_f32_16x16x32_bf16(false, a, false, b, (short)0, c, false, false);
}
__device__ inline float leaky01(float x){ return x >= 0.f ? x : 0.01f*x; }

// ---------------- TDM: async 2D weight-tile fetch into LDS (CDNA5 ISA-8) ----------------
// Loads a (tile_n=64 rows) x (tile_k=32 elems) bf16 tile, row stride = stride_elems, packed
// contiguously into LDS at byte offset lds_off. Tracked by TENSORcnt. Descriptor is
// block-uniform so SGPR groups are legal. 2-group form (2D tensor): VADDR2/3 = NULL.
__device__ inline void tdm_load_b_tile(unsigned int lds_off, const bf16* gp,
                                       unsigned int stride_elems){
  unsigned long long ga = (unsigned long long)(uintptr_t)gp;
  u32x4 g0;
  g0[0] = 1u;                                     // count=1, user descriptor
  g0[1] = lds_off;                                // D#.lds_addr (bytes)
  g0[2] = (unsigned int)ga;                       // global_addr[31:0]
  g0[3] = (unsigned int)((ga >> 32) & 0x1FFFFFFu) // global_addr[56:32]
        | (2u << 30);                             // type=2 ("image")
  u32x8 g1;
  g1[0] = 1u << 16;                               // workgroup_mask=0, data_size=1 (2 bytes)
  g1[1] = 32u << 16;                              // tensor_dim0[15:0] (=32) in bits[63:48]
  g1[2] = 0u | (64u << 16);                       // tensor_dim0[31:16]=0 | tensor_dim1[15:0]=64
  g1[3] = 0u | (32u << 16);                       // tensor_dim1[31:16]=0 | tile_dim0=32
  g1[4] = 64u;                                    // tile_dim1=64 | tile_dim2=0
  g1[5] = stride_elems;                           // tensor_dim0_stride[31:0]
  g1[6] = 0u;                                     // stride[47:32] | dim1_stride[15:0]
  g1[7] = 0u;
  asm volatile("tensor_load_to_lds %0, %1" :: "s"(g0), "s"(g1) : "memory");
}

// ---------------- small utility kernels ----------------
__global__ void f32_to_bf16_kernel(const float* __restrict__ s, bf16* __restrict__ d, int n){
  int i = blockIdx.x*256 + threadIdx.x;
  if (i < n) d[i] = (bf16)s[i];
}

// R (NH,DH,4,DH) [h][d][g][k] -> Rt (NH, 4*DH, DH) row-major (N=g*64+k, K=d) bf16
__global__ void transpose_R_kernel(const float* __restrict__ R, bf16* __restrict__ Rt){
  int i = blockIdx.x*256 + threadIdx.x;
  if (i >= NH_*DH_*4*DH_) return;
  int h   = i / (DH_*4*DH_);
  int rem = i % (DH_*4*DH_);
  int d   = rem / (4*DH_);
  int gk  = rem % (4*DH_);
  Rt[(size_t)h*4*DH_*DH_ + (size_t)gk*DH_ + d] = (bf16)R[i];
}

__global__ void sep_fill_kernel(const float* __restrict__ sep, float* __restrict__ h0){
  int i = blockIdx.x*256 + threadIdx.x;
  if (i >= B_*EMB_) return;
  int b = i / EMB_, d = i % EMB_;
  h0[((size_t)b*S_TOT + 249)*EMB_ + d] = sep[d];
}

// ln_scale: per-token (256-wide) LN, one wave per token; writes f32 (optional) + bf16
__global__ void ln_scale_kernel(const float* __restrict__ X, const float* __restrict__ w,
                                float* __restrict__ outf, bf16* __restrict__ outb, int ntok){
  int lane = threadIdx.x & 31, wave = threadIdx.x >> 5;
  int tk = blockIdx.x*8 + wave;
  if (tk >= ntok) return;
  const float* xr = X + (size_t)tk*EMB_;
  float vals[8], s = 0.f, ss = 0.f;
  #pragma unroll
  for (int i = 0; i < 8; ++i){ float v = xr[lane + 32*i]; vals[i] = v; s += v; ss += v*v; }
  #pragma unroll
  for (int o = 16; o >= 1; o >>= 1){ s += __shfl_xor(s, o, 32); ss += __shfl_xor(ss, o, 32); }
  float mu = s * (1.f/EMB_);
  float rs = rsqrtf(ss * (1.f/EMB_) - mu*mu + 1e-5f);
  #pragma unroll
  for (int i = 0; i < 8; ++i){
    int d = lane + 32*i;
    float o = (vals[i]-mu)*rs*w[d];
    if (outf) outf[(size_t)tk*EMB_ + d] = o;
    outb[(size_t)tk*EMB_ + d] = (bf16)o;
  }
}

// causal depthwise conv (k=3, left pad 2) + silu, bf16 out for gate GEMMs
__global__ void dwconv_silu_kernel(const float* __restrict__ xln, const float* __restrict__ cw,
                                   const float* __restrict__ cb, bf16* __restrict__ xcb){
  int idx = blockIdx.x*256 + threadIdx.x;
  if (idx >= NTOK*EMB_) return;
  int d = idx & 255; int tk = idx >> 8;
  int s = tk % S_TOT, b = tk / S_TOT;
  float acc = cb[d];
  #pragma unroll
  for (int k = 0; k < 3; ++k){
    int ssp = s - 2 + k;
    if (ssp >= 0) acc += xln[((size_t)b*S_TOT + ssp)*EMB_ + d] * cw[d*3 + k];
  }
  float sv = acc / (1.f + expf(-acc));          // silu
  xcb[idx] = (bf16)sv;
}

__global__ void gelu_gate_kernel(const float* __restrict__ u, bf16* __restrict__ gact){
  int idx = blockIdx.x*256 + threadIdx.x;
  if (idx >= NTOK*UPD) return;
  int r = idx / UPD, j = idx % UPD;
  float a = u[(size_t)r*2*UPD + j];
  float g = u[(size_t)r*2*UPD + UPD + j];
  float ge = 0.5f*a*(1.f + erff(a * 0.70710678118f));  // exact gelu
  gact[idx] = (bf16)(ge * g);
}

// ---------------- block-tiled WMMA GEMM with TDM-staged B: C = A(MxK) @ Bw(NxK)^T ----------------
// Grid: (N/64, M/128); 256 threads = 8 waves. Wave w computes rows [w*16, w*16+16) x 64 cols.
// B slab (64 rows x 32 K, bf16, 4KB) is DMA'd by the Tensor Data Mover into LDS, double-buffered,
// shared by all 8 waves; one A fragment feeds 4 WMMAs per K-slab.
__global__ void gemm_bf16_tdm(const bf16* __restrict__ A, int lda,
                              const bf16* __restrict__ Bw, int ldb,
                              float* __restrict__ C, int ldc, int K,
                              const float* __restrict__ bias,
                              const float* __restrict__ Cadd, int ldadd){
  __shared__ __align__(16) bf16 bsm[2][64*32];   // sole LDS alloc -> LDS offset 0
  int lane = threadIdx.x & 31, wave = threadIdx.x >> 5;
  // Visible zero-init of the staging buffers: the real writer is the TDM engine behind
  // opaque asm, and without any visible store the optimizer folds LDS reads to undef.
  *(v8bf*)&bsm[0][threadIdx.x*16]          = (v8bf){0,0,0,0,0,0,0,0};
  *(v8bf*)&bsm[0][2048 + threadIdx.x*16]   = (v8bf){0,0,0,0,0,0,0,0};
  int m0 = blockIdx.y*128 + wave*16;
  int n0 = blockIdx.x*64;
  const bf16* Ap = A  + (size_t)m0*lda;
  const bf16* Bt = Bw + (size_t)n0*ldb;
  int nk = K >> 5;
  v8f acc[4];
  #pragma unroll
  for (int j = 0; j < 4; ++j) acc[j] = (v8f){0.f,0.f,0.f,0.f,0.f,0.f,0.f,0.f};

  __syncthreads();                                // init done before TDM overwrites
  if (wave == 0) tdm_load_b_tile(0u, Bt, (unsigned)ldb);
  for (int kt = 0; kt < nk; ++kt){
    if (wave == 0) __builtin_amdgcn_s_wait_tensorcnt(0);
    __syncthreads();                              // slab kt ready for everyone
    if (wave == 0 && (kt+1) < nk)                 // prefetch next slab into other buffer
      tdm_load_b_tile((unsigned)(((kt+1)&1) * 64*32*2), Bt + (kt+1)*32, (unsigned)ldb);
    if (kt + 2 < nk)                              // light A prefetch
      __builtin_prefetch(Ap + (size_t)(lane & 15)*lda + kt*32 + 64, 0, 1);
    v16bf a = load_a_frag(Ap, lda, kt*32, lane);
    const bf16* Bl = bsm[kt & 1];
    #pragma unroll
    for (int j = 0; j < 4; ++j){
      v16bf b = load_b_frag_lds(Bl + j*16*32, lane);   // ds_load_b128 pair
      acc[j] = wmma_bf16(a, b, acc[j]);
    }
    __syncthreads();                              // done reading before buffer reuse
  }
  int half = lane >> 4;
  #pragma unroll
  for (int j = 0; j < 4; ++j){
    int c = n0 + j*16 + (lane & 15);
    #pragma unroll
    for (int v = 0; v < 8; ++v){
      int r = m0 + v + half*8;
      float val = acc[j][v];
      if (bias) val += bias[c];
      if (Cadd) val += Cadd[(size_t)r*ldadd + c];
      C[(size_t)r*ldc + c] = val;
    }
  }
}

// ---------------- fused tokenizer: conv1/2/3 + LN into LDS, WMMA dense 6656->256 ----------------
__global__ void tokenize_kernel(const float* __restrict__ sig, int P, int s0,
    const float* __restrict__ c1w, const float* __restrict__ c1b,
    const float* __restrict__ l1w, const float* __restrict__ l1b,
    const float* __restrict__ c2w, const float* __restrict__ c2b,
    const float* __restrict__ l2w, const float* __restrict__ l2b,
    const float* __restrict__ c3w, const float* __restrict__ c3b,
    const float* __restrict__ l3w, const float* __restrict__ l3b,
    const bf16* __restrict__ dwb, const float* __restrict__ db,
    float* __restrict__ h0){
  extern __shared__ char smem[];
  bf16*  At   = (bf16*)smem;                       // 16 x LDA_T bf16 feature tile
  float* sg   = (float*)(smem + (size_t)16*LDA_T*2);
  float* h1   = sg  + 64;                          // 32*60
  float* h2s  = h1  + 32*60;                       // 64*56
  float* h3   = h2s + 64*56;                       // 128*52
  float* mu_s = h3  + 128*52;                      // 128
  float* rs_s = mu_s + 128;                        // 128
  int t = threadIdx.x, lane = t & 31, wave = t >> 5;

  for (int pi = 0; pi < 16; ++pi){
    int pid = blockIdx.x*16 + pi;
    int b = pid / P, p = pid % P;
    if (t < 64) sg[t] = sig[((size_t)p*64 + t)*64 + b];   // x = sig.T patch samples
    __syncthreads();
    // conv1 (1->32, k=5) + leaky
    for (int i = t; i < 32*60; i += 256){
      int c = i/60, pos = i%60;
      float a = c1b[c];
      #pragma unroll
      for (int k = 0; k < 5; ++k) a += sg[pos+k]*c1w[c*5+k];
      h1[i] = leaky01(a);
    }
    __syncthreads();
    if (t < 32){
      float s = 0.f, ss = 0.f;
      for (int l = 0; l < 60; ++l){ float v = h1[t*60+l]; s += v; ss += v*v; }
      float mu = s/60.f; mu_s[t] = mu; rs_s[t] = rsqrtf(ss/60.f - mu*mu + 1e-5f);
    }
    __syncthreads();
    for (int i = t; i < 32*60; i += 256){
      int c = i/60, pos = i%60;
      h1[i] = (h1[i]-mu_s[c])*rs_s[c]*l1w[pos] + l1b[pos];
    }
    __syncthreads();
    // conv2 (32->64, k=5) + leaky
    for (int i = t; i < 64*56; i += 256){
      int c = i/56, pos = i%56;
      float a = c2b[c];
      for (int ic = 0; ic < 32; ++ic){
        const float* wp = &c2w[(c*32+ic)*5];
        const float* hp = &h1[ic*60 + pos];
        #pragma unroll
        for (int k = 0; k < 5; ++k) a += hp[k]*wp[k];
      }
      h2s[i] = leaky01(a);
    }
    __syncthreads();
    if (t < 64){
      float s = 0.f, ss = 0.f;
      for (int l = 0; l < 56; ++l){ float v = h2s[t*56+l]; s += v; ss += v*v; }
      float mu = s/56.f; mu_s[t] = mu; rs_s[t] = rsqrtf(ss/56.f - mu*mu + 1e-5f);
    }
    __syncthreads();
    for (int i = t; i < 64*56; i += 256){
      int c = i/56, pos = i%56;
      h2s[i] = (h2s[i]-mu_s[c])*rs_s[c]*l2w[pos] + l2b[pos];
    }
    __syncthreads();
    // conv3 (64->128, k=5) + leaky
    for (int i = t; i < 128*52; i += 256){
      int c = i/52, pos = i%52;
      float a = c3b[c];
      for (int ic = 0; ic < 64; ++ic){
        const float* wp = &c3w[(c*64+ic)*5];
        const float* hp = &h2s[ic*56 + pos];
        #pragma unroll
        for (int k = 0; k < 5; ++k) a += hp[k]*wp[k];
      }
      h3[i] = leaky01(a);
    }
    __syncthreads();
    if (t < 128){
      float s = 0.f, ss = 0.f;
      for (int l = 0; l < 52; ++l){ float v = h3[t*52+l]; s += v; ss += v*v; }
      float mu = s/52.f; mu_s[t] = mu; rs_s[t] = rsqrtf(ss/52.f - mu*mu + 1e-5f);
    }
    __syncthreads();
    for (int i = t; i < 128*52; i += 256){
      int c = i/52, pos = i%52;
      At[(size_t)pi*LDA_T + i] = (bf16)((h3[i]-mu_s[c])*rs_s[c]*l3w[pos] + l3b[pos]);
    }
    __syncthreads();
  }
  // WMMA dense: 16 patches x 6656 features @ dw^T (256,6656) -> tokens
  int nt0 = wave*2;
  v8f acc0 = {0.f,0.f,0.f,0.f,0.f,0.f,0.f,0.f};
  v8f acc1 = {0.f,0.f,0.f,0.f,0.f,0.f,0.f,0.f};
  const bf16* Bp0 = dwb + (size_t)(nt0  )*16*FEAT;
  const bf16* Bp1 = dwb + (size_t)(nt0+1)*16*FEAT;
  for (int kb = 0; kb < FEAT; kb += 32){
    v16bf a  = load_a_frag(At, LDA_T, kb, lane);     // A tile straight from LDS
    v16bf b0 = load_b_frag(Bp0, FEAT, kb, lane);
    acc0 = wmma_bf16(a, b0, acc0);
    v16bf b1 = load_b_frag(Bp1, FEAT, kb, lane);
    acc1 = wmma_bf16(a, b1, acc1);
  }
  int half = lane >> 4;
  int c0 = nt0*16 + (lane & 15);
  #pragma unroll
  for (int v = 0; v < 8; ++v){
    int pid = blockIdx.x*16 + v + half*8;
    int b = pid / P, pp = pid % P;
    size_t row = ((size_t)b*S_TOT + s0 + pp)*EMB_;
    h0[row + c0]      = acc0[v] + db[c0];
    h0[row + c0 + 16] = acc1[v] + db[c0 + 16];
  }
}

// ---------------- sLSTM scan: states in registers, y/r tiles in LDS, WMMA recurrence ----------------
// 4 blocks x 256 threads; block handles 16 batches. Per step: r = y @ Rt^T (per head) via WMMA,
// then the elementwise gate update + head-norm (shfl over the 4 lanes sharing a head row).
__global__ void scan_kernel(const bf16* __restrict__ Rt, const float* __restrict__ Wx,
                            const float* __restrict__ bias, const float* __restrict__ gnw,
                            const float* __restrict__ h0, float* __restrict__ hres){
  extern __shared__ char smem[];
  float* rbuf = (float*)smem;                      // 16 x 1024 f32
  bf16*  ybf  = (bf16*)(smem + (size_t)16*1024*4); // 16 x 256 bf16
  int t = threadIdx.x, lane = t & 31, wave = t >> 5;
  int bblk = blockIdx.x * 16;

  for (int i = t; i < 16*EMB_; i += 256) ybf[i] = (bf16)0.f;

  int lb = t >> 4; int rem = t & 15; int h = rem >> 2; int q = rem & 3; int d0 = q*16;
  int gb = bblk + lb;
  float yv[16], cv[16], nv[16], mv[16];
  #pragma unroll
  for (int j = 0; j < 16; ++j){ yv[j]=0.f; cv[j]=0.f; nv[j]=0.f; mv[j]=0.f; }

  for (int s = 0; s < S_TOT; ++s){
    __syncthreads();                               // ybf ready / rbuf consumed
    #pragma unroll
    for (int i8 = 0; i8 < 8; ++i8){                // 8 waves x 8 tiles = 64 (4 heads x 16 n-tiles)
      int tid = wave*8 + i8;
      int hh = tid >> 4, nt = tid & 15;
      v8f acc = {0.f,0.f,0.f,0.f,0.f,0.f,0.f,0.f};
      const bf16* Ap = ybf + hh*DH_;
      const bf16* Bp = Rt + (size_t)hh*4*DH_*DH_ + (size_t)(nt*16)*DH_;
      #pragma unroll
      for (int kb = 0; kb < DH_; kb += 32){
        v16bf a = load_a_frag(Ap, EMB_, kb, lane);
        v16bf b = load_b_frag(Bp, DH_, kb, lane);
        acc = wmma_bf16(a, b, acc);
      }
      int half = lane >> 4;
      int n = nt*16 + (lane & 15);
      int colr = (n >> 6)*EMB_ + hh*DH_ + (n & 63); // [g][h][k] layout
      #pragma unroll
      for (int v = 0; v < 8; ++v) rbuf[(v + half*8)*1024 + colr] = acc[v];
    }
    __syncthreads();
    size_t wxb = ((size_t)gb * S_TOT + s) * 1024;
    float sum = 0.f, sumsq = 0.f;
    #pragma unroll
    for (int j = 0; j < 16; ++j){
      int ci = h*DH_ + d0 + j;
      float ir  = Wx[wxb +        ci] + rbuf[lb*1024 +        ci] + bias[       ci];
      float fr  = Wx[wxb + 256 +  ci] + rbuf[lb*1024 + 256 +  ci] + bias[256 +  ci];
      float zr  = Wx[wxb + 512 +  ci] + rbuf[lb*1024 + 512 +  ci] + bias[512 +  ci];
      float orr = Wx[wxb + 768 +  ci] + rbuf[lb*1024 + 768 +  ci] + bias[768 +  ci];
      float ls  = fr >= 0.f ? -log1pf(expf(-fr)) : fr - log1pf(expf(fr));  // log_sigmoid
      float lpm = mv[j] + ls;
      float mn  = (nv[j] == 0.f) ? ir : fmaxf(ir, lpm);
      float ig  = expf(ir - mn);
      float fg  = expf(lpm - mn);
      float cn  = fg*cv[j] + ig*tanhf(zr);
      float nn  = fg*nv[j] + ig;
      float sg  = 1.f/(1.f + expf(-orr));
      float yn  = sg*cn/nn;
      yv[j]=yn; cv[j]=cn; nv[j]=nn; mv[j]=mn;
      sum += yn; sumsq += yn*yn;
    }
    // head-norm over DH=64: 4 lanes (q=0..3) share one (b,h) row
    sum   += __shfl_xor(sum,   1, 32); sum   += __shfl_xor(sum,   2, 32);
    sumsq += __shfl_xor(sumsq, 1, 32); sumsq += __shfl_xor(sumsq, 2, 32);
    float mu = sum * (1.f/64.f);
    float rs = rsqrtf(sumsq*(1.f/64.f) - mu*mu + 1e-5f);
    size_t ob = ((size_t)gb * S_TOT + s) * EMB_;
    #pragma unroll
    for (int j = 0; j < 16; ++j){
      int dfull = h*DH_ + d0 + j;
      float o = (yv[j]-mu)*rs*gnw[dfull];
      hres[ob + dfull] = h0[ob + dfull] + o;      // residual add fused
      ybf[lb*EMB_ + dfull] = (bf16)yv[j];
    }
  }
}

// ---------------- head: post-LN on last token, fc1+leaky, fc2 ----------------
__global__ void head_kernel(const float* __restrict__ h2, const float* __restrict__ plw,
                            const float* __restrict__ fc1w, const float* __restrict__ fc1b,
                            const float* __restrict__ fc2w, const float* __restrict__ fc2b,
                            float* __restrict__ out){
  __shared__ float xb[EMB_];
  __shared__ float t1[128];
  __shared__ float st[2];
  int b = blockIdx.x, t = threadIdx.x;
  const float* src = h2 + ((size_t)b*S_TOT + (S_TOT-1))*EMB_;
  xb[t] = src[t]; xb[t+128] = src[t+128];
  __syncthreads();
  if (t == 0){
    float s = 0.f, ss = 0.f;
    for (int k = 0; k < EMB_; ++k){ s += xb[k]; ss += xb[k]*xb[k]; }
    float mu = s*(1.f/EMB_);
    st[0] = mu; st[1] = rsqrtf(ss*(1.f/EMB_) - mu*mu + 1e-5f);
  }
  __syncthreads();
  float mu = st[0], rs = st[1];
  float a0 = (xb[t]-mu)*rs*plw[t];
  float a1 = (xb[t+128]-mu)*rs*plw[t+128];
  __syncthreads();
  xb[t] = a0; xb[t+128] = a1;
  __syncthreads();
  float a = fc1b[t];
  for (int k = 0; k < EMB_; ++k) a += xb[k]*fc1w[t*EMB_ + k];
  t1[t] = leaky01(a);
  __syncthreads();
  if (t < 5){
    float o = fc2b[t];
    for (int k = 0; k < 128; ++k) o += t1[k]*fc2w[t*128 + k];
    out[b*5 + t] = o;
  }
}

// ---------------- host orchestration ----------------
extern "C" void kernel_launch(void* const* d_in, const int* in_sizes, int n_in,
                              void* d_out, int out_size, void* d_ws, size_t ws_size,
                              hipStream_t stream){
  (void)in_sizes; (void)n_in; (void)out_size; (void)ws_size;
  const float* ctx = (const float*)d_in[0];
  const float* xx  = (const float*)d_in[1];
  const float* c1w = (const float*)d_in[2];
  const float* c1b = (const float*)d_in[3];
  const float* l1w = (const float*)d_in[4];
  const float* l1b = (const float*)d_in[5];
  const float* c2w = (const float*)d_in[6];
  const float* c2b = (const float*)d_in[7];
  const float* l2w = (const float*)d_in[8];
  const float* l2b = (const float*)d_in[9];
  const float* c3w = (const float*)d_in[10];
  const float* c3b = (const float*)d_in[11];
  const float* l3w = (const float*)d_in[12];
  const float* l3b = (const float*)d_in[13];
  const float* dw  = (const float*)d_in[14];
  const float* db  = (const float*)d_in[15];
  const float* sep = (const float*)d_in[16];
  const float* blk_ln_w = (const float*)d_in[17];
  const float* cw  = (const float*)d_in[18];
  const float* cb  = (const float*)d_in[19];
  const float* iw  = (const float*)d_in[20];
  const float* fwt = (const float*)d_in[21];
  const float* zw  = (const float*)d_in[22];
  const float* owt = (const float*)d_in[23];
  const float* Rw  = (const float*)d_in[24];
  const float* bias= (const float*)d_in[25];
  const float* gnw = (const float*)d_in[26];
  const float* ffn_ln_w = (const float*)d_in[27];
  const float* upw = (const float*)d_in[28];
  const float* dnw = (const float*)d_in[29];
  const float* post_ln_w = (const float*)d_in[30];
  const float* fc1w = (const float*)d_in[31];
  const float* fc1b = (const float*)d_in[32];
  const float* fc2w = (const float*)d_in[33];
  const float* fc2b = (const float*)d_in[34];
  float* out = (float*)d_out;

  char* wsp = (char*)d_ws; size_t off = 0;
  auto take = [&](size_t bytes)->void*{
    void* p = wsp + off; off += (bytes + 255) & ~(size_t)255; return p;
  };
  bf16*  dwb  = (bf16*)take((size_t)EMB_*FEAT*2);
  bf16*  upwb = (bf16*)take((size_t)2*UPD*EMB_*2);
  bf16*  dnwb = (bf16*)take((size_t)EMB_*UPD*2);
  bf16*  gwb  = (bf16*)take((size_t)4*NH_*DH_*DH_*2);   // [g][h][o][d], g: i,f,z,o
  bf16*  Rtb  = (bf16*)take((size_t)NH_*4*DH_*DH_*2);
  float* h0   = (float*)take((size_t)NTOK*EMB_*4);
  float* xlnf = (float*)take((size_t)NTOK*EMB_*4);
  bf16*  xlnb = (bf16*)take((size_t)NTOK*EMB_*2);
  bf16*  xcb  = (bf16*)take((size_t)NTOK*EMB_*2);
  float* Wx   = (float*)take((size_t)NTOK*4*EMB_*4);    // [b][s][g][h][o]
  float* hres = (float*)take((size_t)NTOK*EMB_*4);
  bf16*  flnb = (bf16*)take((size_t)NTOK*EMB_*2);
  float* u    = (float*)take((size_t)NTOK*2*UPD*4);
  bf16*  gact = (bf16*)take((size_t)NTOK*UPD*2);
  float* h2   = (float*)take((size_t)NTOK*EMB_*4);

  auto cvt = [&](const float* s, bf16* d, int n){
    f32_to_bf16_kernel<<<(n+255)/256, 256, 0, stream>>>(s, d, n);
  };
  cvt(dw,  dwb,  EMB_*FEAT);
  cvt(upw, upwb, 2*UPD*EMB_);
  cvt(dnw, dnwb, EMB_*UPD);
  cvt(iw,  gwb + 0*NH_*DH_*DH_, NH_*DH_*DH_);
  cvt(fwt, gwb + 1*NH_*DH_*DH_, NH_*DH_*DH_);
  cvt(zw,  gwb + 2*NH_*DH_*DH_, NH_*DH_*DH_);
  cvt(owt, gwb + 3*NH_*DH_*DH_, NH_*DH_*DH_);
  transpose_R_kernel<<<(NH_*DH_*4*DH_+255)/256, 256, 0, stream>>>(Rw, Rtb);

  size_t tok_smem = (size_t)16*LDA_T*2 + (size_t)(64 + 32*60 + 64*56 + 128*52 + 256)*4;
  tokenize_kernel<<<996, 256, tok_smem, stream>>>(ctx, 249, 0,
      c1w,c1b,l1w,l1b, c2w,c2b,l2w,l2b, c3w,c3b,l3w,l3b, dwb, db, h0);
  tokenize_kernel<<<1000, 256, tok_smem, stream>>>(xx, 250, 250,
      c1w,c1b,l1w,l1b, c2w,c2b,l2w,l2b, c3w,c3b,l3w,l3b, dwb, db, h0);
  sep_fill_kernel<<<(B_*EMB_+255)/256, 256, 0, stream>>>(sep, h0);

  ln_scale_kernel<<<(NTOK+7)/8, 256, 0, stream>>>(h0, blk_ln_w, xlnf, xlnb, NTOK);
  dwconv_silu_kernel<<<(NTOK*EMB_+255)/256, 256, 0, stream>>>(xlnf, cw, cb, xcb);

  // gate projections: 4 gates x 4 heads, M=32000 N=64 K=64 each (TDM-staged B)
  for (int gg = 0; gg < 4; ++gg){
    const bf16* Asrc = (gg < 2) ? xcb : xlnb;    // i,f from conv'd path; z,o from ln(x)
    for (int h = 0; h < NH_; ++h){
      dim3 grid(1, NTOK/128);
      gemm_bf16_tdm<<<grid, 256, 0, stream>>>(
          Asrc + h*DH_, EMB_,
          gwb + (size_t)(gg*NH_ + h)*DH_*DH_, DH_,
          Wx + gg*EMB_ + h*DH_, 4*EMB_,
          DH_, nullptr, nullptr, 0);
    }
  }

  size_t scan_smem = (size_t)16*1024*4 + (size_t)16*EMB_*2;
  scan_kernel<<<4, 256, scan_smem, stream>>>(Rtb, Wx, bias, gnw, h0, hres);

  ln_scale_kernel<<<(NTOK+7)/8, 256, 0, stream>>>(hres, ffn_ln_w, nullptr, flnb, NTOK);
  {
    dim3 grid((2*UPD)/64, NTOK/128);
    gemm_bf16_tdm<<<grid, 256, 0, stream>>>(flnb, EMB_, upwb, EMB_,
        u, 2*UPD, EMB_, nullptr, nullptr, 0);
  }
  gelu_gate_kernel<<<(NTOK*UPD+255)/256, 256, 0, stream>>>(u, gact);
  {
    dim3 grid(EMB_/64, NTOK/128);
    gemm_bf16_tdm<<<grid, 256, 0, stream>>>(gact, UPD, dnwb, UPD,
        h2, EMB_, UPD, nullptr, hres, EMB_);     // residual fused
  }
  head_kernel<<<B_, 128, 0, stream>>>(h2, post_ln_w, fc1w, fc1b, fc2w, fc2b, out);
}
